// Qwen2MoTDecoderLayer_1211180777543
// MI455X (gfx1250) — matrix operand order repaired and checked by reading, not compiled, not run on hardware
//
#include <hip/hip_runtime.h>
#include <hip/hip_bf16.h>

typedef __bf16 bf16;
typedef __attribute__((ext_vector_type(16))) __bf16 v16bf;
typedef __attribute__((ext_vector_type(8)))  float  v8f;

#define DEVINL __device__ __forceinline__

constexpr int T_TOK = 4096;
constexpr int HID   = 2048;
constexpr int NH    = 16;
constexpr int NKV   = 8;
constexpr int HD    = 128;
constexpr int INTER = 5632;
constexpr int Bn    = 4;
constexpr int Sn    = 1024;

#if defined(__has_builtin)
# if __has_builtin(__builtin_amdgcn_tensor_load_to_lds) && __has_builtin(__builtin_amdgcn_s_wait_tensorcnt)
#  define USE_TDM 1
# endif
#endif
#ifndef USE_TDM
# define USE_TDM 0
#endif

union FragU { v16bf v; unsigned u[8]; };

DEVINL v8f v8f_zero() { v8f z = {0.f,0.f,0.f,0.f,0.f,0.f,0.f,0.f}; return z; }

// A-matrix 16x32 bf16 fragment (contiguous k order, matching the B loads): lane m = lane&15 selects row,
// VGPR j holds K pair: K = hi*16 + 2j
DEVINL v16bf load_a_frag(const bf16* row, int k0, int hi) {
  FragU a;
#pragma unroll
  for (int j = 0; j < 8; ++j) {
    int kk = hi * 16 + 2 * j;
    a.u[j] = *reinterpret_cast<const unsigned*>(row + k0 + kk);
  }
  return a.v;
}

// B-matrix 32x16 bf16 fragment from K-contiguous storage (row = one N column,
// K contiguous): lane n = lane&15, K pair = hi*16 + 2j.
DEVINL v16bf load_b_frag(const bf16* row, int k0, int hi) {
  FragU b;
#pragma unroll
  for (int j = 0; j < 8; ++j)
    b.u[j] = *reinterpret_cast<const unsigned*>(row + k0 + hi * 16 + 2 * j);
  return b.v;
}

#if USE_TDM
typedef unsigned v4u __attribute__((ext_vector_type(4)));
typedef int      v8i __attribute__((ext_vector_type(8)));
typedef int      v4i __attribute__((ext_vector_type(4)));

// 2-D tile TDM load, bf16 elements. Builds D# group0/group1 per ISA 08 §8.3/8.4:
// group0: count=1, lds_addr, 57-bit global tile addr, type=2 ("image").
// group1: data_size=2B, tensor dims == tile dims (no OOB), tile_dim0/1,
//         tensor_dim0_stride in elements; everything else zero.
DEVINL void tdm_load_2d(unsigned ldsOff, const void* gptr,
                        unsigned tileD0, unsigned tileD1, unsigned strideElems) {
  unsigned long long ga = (unsigned long long)(size_t)gptr;
  v4u g0;
  g0.x = 1u;                                                    // count=1 (user)
  g0.y = ldsOff;                                                // lds_addr (bytes)
  g0.z = (unsigned)ga;                                          // global_addr[31:0]
  g0.w = (unsigned)((ga >> 32) & 0x01FFFFFFu) | (2u << 30);     // addr[56:32] | type=2
  v8i g1;
  g1[0] = (int)(1u << 16);                                      // data_size=1 -> 2 bytes
  g1[1] = (int)((tileD0 & 0xFFFFu) << 16);                      // tensor_dim0[15:0] @ bit48
  g1[2] = (int)((tileD0 >> 16) | ((tileD1 & 0xFFFFu) << 16));   // tdim0 hi | tensor_dim1 lo
  g1[3] = (int)((tileD1 >> 16) | ((tileD0 & 0xFFFFu) << 16));   // tdim1 hi | tile_dim0
  g1[4] = (int)(tileD1 & 0xFFFFu);                              // tile_dim1 (tile_dim2=0)
  g1[5] = (int)strideElems;                                     // tensor_dim0_stride[31:0]
  g1[6] = 0;
  g1[7] = 0;
  v4i z4 = {0, 0, 0, 0};
#if __clang_major__ >= 23
  v8i z8 = {0, 0, 0, 0, 0, 0, 0, 0};
  __builtin_amdgcn_tensor_load_to_lds(g0, g1, z4, z4, z8, 0);
#else
  __builtin_amdgcn_tensor_load_to_lds(g0, g1, z4, z4, 0);
#endif
}
#endif // USE_TDM

// ---------------- weight prep: fp32 [K,N] -> bf16 transposed [N,K] -------------
__global__ __launch_bounds__(256)
void k_cvt_transpose(const float* __restrict__ src, bf16* __restrict__ dst,
                     int K, int N) {
  size_t idx = (size_t)blockIdx.x * 256 + threadIdx.x;
  size_t total = (size_t)K * N;
  if (idx >= total) return;
  int k = (int)(idx / N), n = (int)(idx % N);
  dst[(size_t)n * K + k] = (bf16)src[idx];
}

// ---------------- routed RMSNorm: fp32 in -> bf16 out --------------------------
__global__ __launch_bounds__(256)
void k_rmsnorm_parity(const float* __restrict__ x, const float* __restrict__ w0,
                      const float* __restrict__ w1, bf16* __restrict__ out, int H) {
  const int t = blockIdx.x;
  const float* w = (t & 1) ? w1 : w0;
  const float* xr = x + (size_t)t * H;
  float ss = 0.f;
  for (int i = threadIdx.x; i < H; i += 256) { float v = xr[i]; ss += v * v; }
#pragma unroll
  for (int off = 16; off >= 1; off >>= 1) ss += __shfl_xor(ss, off, 32);
  __shared__ float red[8];
  if ((threadIdx.x & 31) == 0) red[threadIdx.x >> 5] = ss;
  __syncthreads();
  float tot = 0.f;
#pragma unroll
  for (int wv = 0; wv < 8; ++wv) tot += red[wv];
  float rinv = rsqrtf(tot / (float)H + 1e-6f);
  bf16* orow = out + (size_t)t * H;
  for (int i = threadIdx.x; i < H; i += 256) orow[i] = (bf16)(xr[i] * rinv * w[i]);
}

// ---------------- parity-routed GEMM on bf16 WMMA ------------------------------
// 32x64 output tile per wave: 2 A frags x 4 B frags -> 8 independent WMMAs per
// K-step (keeps the XDL pipe saturated), ~21 flops/byte of fetched operands.
// grid = (N/64, (T/2)/32, 2(parity)); block = 1 wave.
template <bool OUT_BF16>
__global__ __launch_bounds__(32)
void k_gemm_parity(const bf16* __restrict__ A, const bf16* __restrict__ W0,
                   const bf16* __restrict__ W1, const float* __restrict__ b0,
                   const float* __restrict__ b1, const float* __restrict__ resid,
                   float* __restrict__ outF, bf16* __restrict__ outB,
                   int K, int N) {
  const int lane = threadIdx.x;
  const int m  = lane & 15;
  const int hi = lane >> 4;
  const int parity = blockIdx.z;
  const bf16* Wt    = parity ? W1 : W0;
  const float* bias = parity ? b1 : b0;
  const int n0 = (int)blockIdx.x * 64;
  const int r0 = (int)blockIdx.y * 32;

  const bf16* Ar0 = A + (size_t)(2 * (r0 + m) + parity) * K;
  const bf16* Ar1 = A + (size_t)(2 * (r0 + 16 + m) + parity) * K;
  const bf16* Br0 = Wt + (size_t)(n0 + m) * K;
  const bf16* Br1 = Wt + (size_t)(n0 + 16 + m) * K;
  const bf16* Br2 = Wt + (size_t)(n0 + 32 + m) * K;
  const bf16* Br3 = Wt + (size_t)(n0 + 48 + m) * K;

  v8f acc[2][4];
#pragma unroll
  for (int mi = 0; mi < 2; ++mi)
#pragma unroll
    for (int nb = 0; nb < 4; ++nb) acc[mi][nb] = v8f_zero();

  for (int k0 = 0; k0 < K; k0 += 32) {
    v16bf a0 = load_a_frag(Ar0, k0, hi);
    v16bf a1 = load_a_frag(Ar1, k0, hi);
    v16bf bb0 = load_b_frag(Br0, k0, hi);
    v16bf bb1 = load_b_frag(Br1, k0, hi);
    v16bf bb2 = load_b_frag(Br2, k0, hi);
    v16bf bb3 = load_b_frag(Br3, k0, hi);
    acc[0][0] = __builtin_amdgcn_wmma_f32_16x16x32_bf16(false, a0, false, bb0, (short)0, acc[0][0], false, false);
    acc[1][0] = __builtin_amdgcn_wmma_f32_16x16x32_bf16(false, a1, false, bb0, (short)0, acc[1][0], false, false);
    acc[0][1] = __builtin_amdgcn_wmma_f32_16x16x32_bf16(false, a0, false, bb1, (short)0, acc[0][1], false, false);
    acc[1][1] = __builtin_amdgcn_wmma_f32_16x16x32_bf16(false, a1, false, bb1, (short)0, acc[1][1], false, false);
    acc[0][2] = __builtin_amdgcn_wmma_f32_16x16x32_bf16(false, a0, false, bb2, (short)0, acc[0][2], false, false);
    acc[1][2] = __builtin_amdgcn_wmma_f32_16x16x32_bf16(false, a1, false, bb2, (short)0, acc[1][2], false, false);
    acc[0][3] = __builtin_amdgcn_wmma_f32_16x16x32_bf16(false, a0, false, bb3, (short)0, acc[0][3], false, false);
    acc[1][3] = __builtin_amdgcn_wmma_f32_16x16x32_bf16(false, a1, false, bb3, (short)0, acc[1][3], false, false);
  }

  float bv[4];
#pragma unroll
  for (int nb = 0; nb < 4; ++nb) bv[nb] = bias ? bias[n0 + nb * 16 + m] : 0.f;

#pragma unroll
  for (int mi = 0; mi < 2; ++mi) {
#pragma unroll
    for (int j = 0; j < 8; ++j) {
      int row = r0 + mi * 16 + j + hi * 8;
      int tok = 2 * row + parity;
#pragma unroll
      for (int nb = 0; nb < 4; ++nb) {
        size_t off = (size_t)tok * N + n0 + nb * 16 + m;
        float v = acc[mi][nb][j] + bv[nb];
        if (resid) v += resid[off];
        if (OUT_BF16) outB[off] = (bf16)v; else outF[off] = v;
      }
    }
  }
}

// ---------------- RoPE + per-head routed RMSNorm: fp32 in -> bf16 out ----------
__global__ __launch_bounds__(128)
void k_rope_rms(const float* __restrict__ qin, const float* __restrict__ cosp,
                const float* __restrict__ sinp, const float* __restrict__ w0,
                const float* __restrict__ w1, bf16* __restrict__ outb, int NHx) {
  const int t = blockIdx.x / NHx;
  const int h = blockIdx.x % NHx;
  const int d = threadIdx.x; // 0..127
  size_t base = ((size_t)t * NHx + h) * HD;
  float q   = qin[base + d];
  float rot = (d < 64) ? -qin[base + d + 64] : qin[base + d - 64];
  float val = q * cosp[(size_t)t * HD + d] + rot * sinp[(size_t)t * HD + d];
  float ss = val * val;
#pragma unroll
  for (int off = 16; off >= 1; off >>= 1) ss += __shfl_xor(ss, off, 32);
  __shared__ float red[4];
  if ((threadIdx.x & 31) == 0) red[threadIdx.x >> 5] = ss;
  __syncthreads();
  float tot = red[0] + red[1] + red[2] + red[3];
  float rinv = rsqrtf(tot / (float)HD + 1e-6f);
  const float* w = (t & 1) ? w1 : w0;
  outb[base + d] = (bf16)(val * rinv * w[d]);
}

// ---------------- V: fp32 [T,NKV,HD] -> bf16 seq-minor [B,NKV,HD,Sn] -----------
__global__ __launch_bounds__(256)
void k_v_prep(const float* __restrict__ vf, bf16* __restrict__ vt) {
  size_t idx = (size_t)blockIdx.x * 256 + threadIdx.x;
  if (idx >= (size_t)T_TOK * NKV * HD) return;
  int t = (int)(idx / (NKV * HD));
  int c = (int)(idx % (NKV * HD));
  int kv = c / HD, d = c % HD;
  int b = t / Sn, s = t % Sn;
  vt[((size_t)(b * NKV + kv) * HD + d) * Sn + s] = (bf16)vf[idx];
}

// ---------------- flash attention --------------------------------------------
// One block = the two query heads of one GQA group (2 waves) over a 16-query
// tile. K/V tiles are shared in LDS, DMA'd by the Tensor Data Mover with
// double buffering: wave 0 issues block i+1's two TDM loads into the ping-pong
// buffer, then s_wait_tensorcnt 2 (TENSORcnt completes in order, so block i's
// pair is guaranteed resident) while the DMA overlaps the WMMA work; a
// workgroup barrier publishes the tiles to wave 1. Falls back to direct global
// loads if the TDM builtin is unavailable.
__global__ __launch_bounds__(64)
void k_attention(const bf16* __restrict__ qb, const bf16* __restrict__ kb,
                 const bf16* __restrict__ vt, bf16* __restrict__ ob) {
  const int lane = threadIdx.x & 31;
  const int w    = threadIdx.x >> 5;   // wave in block = head within GQA group
  const int m  = lane & 15;
  const int hi = lane >> 4;
  const int qt = blockIdx.x;           // 16-query tile
  const int kv = blockIdx.y;           // kv head
  const int b  = blockIdx.z;
  const int h  = kv * 2 + w;           // query head
  const float scale = 0.08838834764831845f; // 1/sqrt(128)

  __shared__ bf16 Plds[2][16 * 32];    // per-wave P transpose staging
#if USE_TDM
  __shared__ bf16 Ktile[2][32 * HD];   // ping-pong [key][dim]
  __shared__ bf16 Vtile[2][HD * 32];   // ping-pong [dim][key]
  const bf16* kgbase = kb + (size_t)b * Sn * NKV * HD + (size_t)kv * HD;
  const bf16* vgbase = vt + (size_t)(b * NKV + kv) * HD * Sn;
#endif
  bf16* pl = Plds[w];

  // resident Q fragments (16 rows x 128 dims = 4 x K32 fragments)
  const bf16* qrow = qb + ((size_t)(b * Sn + qt * 16 + m) * NH + h) * HD;
  v16bf qf[4];
#pragma unroll
  for (int kk = 0; kk < 4; ++kk) qf[kk] = load_a_frag(qrow, kk * 32, hi);

  v8f o[8];
#pragma unroll
  for (int d = 0; d < 8; ++d) o[d] = v8f_zero();
  float mrun[8], lrun[8];
#pragma unroll
  for (int j = 0; j < 8; ++j) { mrun[j] = -1e30f; lrun[j] = 0.f; }

  const int nblk = (qt * 16 + 47) >> 5; // 32-key blocks up to the causal frontier

#if USE_TDM
  if (w == 0) { // prologue: DMA block 0 into buffer 0
    tdm_load_2d((unsigned)(size_t)&Ktile[0][0], kgbase, HD, 32, NKV * HD);
    tdm_load_2d((unsigned)(size_t)&Vtile[0][0], vgbase, 32, HD, Sn);
  }
#endif

  for (int blk = 0; blk < nblk; ++blk) {
    const int s0 = blk * 32;
    const int cur = blk & 1;

#if USE_TDM
    if (w == 0) {
      if (blk + 1 < nblk) { // DMA next block into the other buffer, overlap
        const int s1 = s0 + 32;
        tdm_load_2d((unsigned)(size_t)&Ktile[1 - cur][0],
                    kgbase + (size_t)s1 * NKV * HD, HD, 32, NKV * HD);
        tdm_load_2d((unsigned)(size_t)&Vtile[1 - cur][0],
                    vgbase + s1, 32, HD, Sn);
        __builtin_amdgcn_s_wait_tensorcnt((short)2); // block blk's pair landed
      } else {
        __builtin_amdgcn_s_wait_tensorcnt((short)0);
      }
    }
    __syncthreads(); // tiles for blk visible to both waves
#endif

    float sc[2][8];
#pragma unroll
    for (int nt = 0; nt < 2; ++nt) {
#if USE_TDM
      const bf16* krow = &Ktile[cur][(size_t)(nt * 16 + m) * HD];
#else
      const bf16* krow = kb + ((size_t)(b * Sn + s0 + nt * 16 + m) * NKV + kv) * HD;
#endif
      v8f s = v8f_zero();
#pragma unroll
      for (int kk = 0; kk < 4; ++kk) {
        v16bf bfr = load_b_frag(krow, kk * 32, hi);
        s = __builtin_amdgcn_wmma_f32_16x16x32_bf16(false, qf[kk], false, bfr,
                                                    (short)0, s, false, false);
      }
#pragma unroll
      for (int j = 0; j < 8; ++j) {
        int qr = qt * 16 + j + hi * 8;
        int kc = s0 + nt * 16 + m;
        sc[nt][j] = (kc <= qr) ? s[j] * scale : -1e30f;
      }
    }

    // online softmax; each C-row lives in one lane-half at one VGPR index,
    // so 16-lane xor-shuffles reduce each row and stats map 1:1 onto O VGPRs.
#pragma unroll
    for (int j = 0; j < 8; ++j) {
      float rm = fmaxf(sc[0][j], sc[1][j]);
#pragma unroll
      for (int off = 8; off >= 1; off >>= 1) rm = fmaxf(rm, __shfl_xor(rm, off, 32));
      float mn = fmaxf(mrun[j], rm);
      float f  = __expf(mrun[j] - mn);
      float p0 = __expf(sc[0][j] - mn);
      float p1 = __expf(sc[1][j] - mn);
      float rs = p0 + p1;
#pragma unroll
      for (int off = 8; off >= 1; off >>= 1) rs += __shfl_xor(rs, off, 32);
      lrun[j] = lrun[j] * f + rs;
      mrun[j] = mn;
#pragma unroll
      for (int d = 0; d < 8; ++d) o[d][j] *= f;
      pl[(j + hi * 8) * 32 + m]      = (bf16)p0;
      pl[(j + hi * 8) * 32 + 16 + m] = (bf16)p1;
    }
    __syncthreads(); // P staged (also keeps both waves in lockstep)
    v16bf pa = load_a_frag(&pl[m * 32], 0, hi);
#pragma unroll
    for (int d = 0; d < 8; ++d) {
      FragU bb;
#if USE_TDM
      const bf16* vrow = &Vtile[cur][(size_t)(d * 16 + m) * 32];
#pragma unroll
      for (int j = 0; j < 8; ++j)
        bb.u[j] = *reinterpret_cast<const unsigned*>(vrow + hi * 16 + 2 * j);
#else
      const bf16* vrow = vt + ((size_t)((b * NKV + kv) * HD + d * 16 + m)) * Sn;
#pragma unroll
      for (int j = 0; j < 8; ++j)
        bb.u[j] = *reinterpret_cast<const unsigned*>(vrow + s0 + hi * 16 + 2 * j);
#endif
      o[d] = __builtin_amdgcn_wmma_f32_16x16x32_bf16(false, pa, false, bb.v,
                                                     (short)0, o[d], false, false);
    }
    __syncthreads(); // both waves done with buf[cur] / P before reuse
  }

#pragma unroll
  for (int j = 0; j < 8; ++j) {
    float inv = 1.f / lrun[j];
    int tok = b * Sn + qt * 16 + j + hi * 8;
#pragma unroll
    for (int d = 0; d < 8; ++d)
      ob[((size_t)tok * NH + h) * HD + d * 16 + m] = (bf16)(o[d][j] * inv);
  }
}

// ---------------- SwiGLU elementwise: m = silu(g) * u --------------------------
__global__ __launch_bounds__(256)
void k_silu_mul(const bf16* __restrict__ g, const bf16* __restrict__ u,
                bf16* __restrict__ mout) {
  size_t idx = (size_t)blockIdx.x * 256 + threadIdx.x;
  if (idx >= (size_t)T_TOK * INTER) return;
  float gv = (float)g[idx];
  float uv = (float)u[idx];
  mout[idx] = (bf16)(gv / (1.f + __expf(-gv)) * uv);
}

// ==============================================================================
extern "C" void kernel_launch(void* const* d_in, const int* in_sizes, int n_in,
                              void* d_out, int out_size, void* d_ws, size_t ws_size,
                              hipStream_t stream) {
  (void)in_sizes; (void)n_in; (void)out_size; (void)ws_size;
  const float* x    = (const float*)d_in[0];
  const float* cosp = (const float*)d_in[2];
  const float* sinp = (const float*)d_in[3];
  enum { LN_IN, LN_IN_G, LN_POST, LN_POST_G, QN, KN, QN_G, KN_G,
         WQ, BQ, WK, BK, WV, BV, WO,
         WQ_G, BQ_G, WK_G, BK_G, WV_G, BV_G, WO_G,
         GATE, UP, DOWN, GATE_G, UP_G, DOWN_G };
  const float* P[28];
  for (int i = 0; i < 28; ++i) P[i] = (const float*)d_in[6 + i];

  char* wsb = (char*)d_ws;
  size_t off = 0;
  auto alloc = [&](size_t bytes) {
    void* p = wsb + off;
    off += (bytes + 255) & ~(size_t)255;
    return p;
  };

  // bf16 transposed weights [N,K]
  const size_t SZ_2048x2048 = (size_t)2048 * 2048 * 2;
  const size_t SZ_1024x2048 = (size_t)1024 * 2048 * 2;
  const size_t SZ_5632x2048 = (size_t)5632 * 2048 * 2;
  bf16* wqt[2]   = { (bf16*)alloc(SZ_2048x2048), (bf16*)alloc(SZ_2048x2048) };
  bf16* wkt[2]   = { (bf16*)alloc(SZ_1024x2048), (bf16*)alloc(SZ_1024x2048) };
  bf16* wvt[2]   = { (bf16*)alloc(SZ_1024x2048), (bf16*)alloc(SZ_1024x2048) };
  bf16* wot[2]   = { (bf16*)alloc(SZ_2048x2048), (bf16*)alloc(SZ_2048x2048) };
  bf16* gatet[2] = { (bf16*)alloc(SZ_5632x2048), (bf16*)alloc(SZ_5632x2048) };
  bf16* upt[2]   = { (bf16*)alloc(SZ_5632x2048), (bf16*)alloc(SZ_5632x2048) };
  bf16* downt[2] = { (bf16*)alloc(SZ_5632x2048), (bf16*)alloc(SZ_5632x2048) };

  // activations
  bf16*  h_b = (bf16*)alloc((size_t)T_TOK * HID * 2);
  float* x1  = (float*)alloc((size_t)T_TOK * HID * 4);
  bf16*  o_b = (bf16*)alloc((size_t)T_TOK * HID * 2);
  bf16*  q_b = (bf16*)alloc((size_t)T_TOK * NH * HD * 2);
  bf16*  k_b = (bf16*)alloc((size_t)T_TOK * NKV * HD * 2);
  bf16*  v_t = (bf16*)alloc((size_t)T_TOK * NKV * HD * 2);
  // big region: first holds fp32 q/k/v, later reused for bf16 gate/up/m
  char*  big = (char*)alloc(2 * (size_t)T_TOK * INTER * 2);
  float* qf = (float*)big;
  float* kf = (float*)(big + (size_t)T_TOK * NH * HD * 4);
  float* vf = (float*)(big + (size_t)T_TOK * NH * HD * 4 + (size_t)T_TOK * NKV * HD * 4);
  bf16*  g_b = (bf16*)big;
  bf16*  u_b = (bf16*)(big + (size_t)T_TOK * INTER * 2);
  bf16*  m_b = g_b; // in-place silu*up

  auto cvt = [&](const float* src, bf16* dst, int K, int N) {
    int blocks = (int)(((size_t)K * N + 255) / 256);
    k_cvt_transpose<<<blocks, 256, 0, stream>>>(src, dst, K, N);
  };
  cvt(P[WQ],     wqt[0],   HID,   NH * HD);
  cvt(P[WQ_G],   wqt[1],   HID,   NH * HD);
  cvt(P[WK],     wkt[0],   HID,   NKV * HD);
  cvt(P[WK_G],   wkt[1],   HID,   NKV * HD);
  cvt(P[WV],     wvt[0],   HID,   NKV * HD);
  cvt(P[WV_G],   wvt[1],   HID,   NKV * HD);
  cvt(P[WO],     wot[0],   NH * HD, HID);
  cvt(P[WO_G],   wot[1],   NH * HD, HID);
  cvt(P[GATE],   gatet[0], HID,   INTER);
  cvt(P[GATE_G], gatet[1], HID,   INTER);
  cvt(P[UP],     upt[0],   HID,   INTER);
  cvt(P[UP_G],   upt[1],   HID,   INTER);
  cvt(P[DOWN],   downt[0], INTER, HID);
  cvt(P[DOWN_G], downt[1], INTER, HID);

  // 1) input RMSNorm (routed by parity)
  k_rmsnorm_parity<<<T_TOK, 256, 0, stream>>>(x, P[LN_IN], P[LN_IN_G], h_b, HID);

  // 2) QKV projections (parity-routed bf16 WMMA GEMMs, fp32 out for RoPE)
  dim3 gq(NH * HD / 64, (T_TOK / 2) / 32, 2);
  dim3 gk(NKV * HD / 64, (T_TOK / 2) / 32, 2);
  k_gemm_parity<false><<<gq, 32, 0, stream>>>(h_b, wqt[0], wqt[1], P[BQ], P[BQ_G],
                                              nullptr, qf, nullptr, HID, NH * HD);
  k_gemm_parity<false><<<gk, 32, 0, stream>>>(h_b, wkt[0], wkt[1], P[BK], P[BK_G],
                                              nullptr, kf, nullptr, HID, NKV * HD);
  k_gemm_parity<false><<<gk, 32, 0, stream>>>(h_b, wvt[0], wvt[1], P[BV], P[BV_G],
                                              nullptr, vf, nullptr, HID, NKV * HD);

  // 3) RoPE + routed q/k head RMSNorm -> bf16
  k_rope_rms<<<T_TOK * NH, 128, 0, stream>>>(qf, cosp, sinp, P[QN], P[QN_G], q_b, NH);
  k_rope_rms<<<T_TOK * NKV, 128, 0, stream>>>(kf, cosp, sinp, P[KN], P[KN_G], k_b, NKV);
  k_v_prep<<<(int)(((size_t)T_TOK * NKV * HD + 255) / 256), 256, 0, stream>>>(vf, v_t);

  // 4) causal GQA flash attention: 2 q-heads per block share double-buffered
  //    TDM-staged K/V tiles; WMMA QK^T / PV -> bf16
  dim3 ga(Sn / 16, NKV, Bn);
  k_attention<<<ga, 64, 0, stream>>>(q_b, k_b, v_t, o_b);

  // 5) O projection with fused residual add -> x1 (fp32)
  dim3 go(HID / 64, (T_TOK / 2) / 32, 2);
  k_gemm_parity<false><<<go, 32, 0, stream>>>(o_b, wot[0], wot[1], nullptr, nullptr,
                                              x, x1, nullptr, NH * HD, HID);

  // 6) post RMSNorm (routed) -> reuse h_b
  k_rmsnorm_parity<<<T_TOK, 256, 0, stream>>>(x1, P[LN_POST], P[LN_POST_G], h_b, HID);

  // 7) SwiGLU MLP: gate / up GEMMs (bf16 out), silu*mul, down GEMM (+residual -> d_out)
  dim3 gm(INTER / 64, (T_TOK / 2) / 32, 2);
  k_gemm_parity<true><<<gm, 32, 0, stream>>>(h_b, gatet[0], gatet[1], nullptr, nullptr,
                                             nullptr, nullptr, g_b, HID, INTER);
  k_gemm_parity<true><<<gm, 32, 0, stream>>>(h_b, upt[0], upt[1], nullptr, nullptr,
                                             nullptr, nullptr, u_b, HID, INTER);
  k_silu_mul<<<(int)(((size_t)T_TOK * INTER + 255) / 256), 256, 0, stream>>>(g_b, u_b, m_b);

  dim3 gd(HID / 64, (T_TOK / 2) / 32, 2);
  k_gemm_parity<false><<<gd, 32, 0, stream>>>(m_b, downt[0], downt[1], nullptr, nullptr,
                                              x1, (float*)d_out, nullptr, INTER, HID);
}
